// TSP_Decoder_3358664425904
// MI455X (gfx1250) — matrix operand order repaired and checked by reading, not compiled, not run on hardware
//
#include <hip/hip_runtime.h>

// ---------------------------------------------------------------------------
// MI455X (gfx1250, wave32) implementation of the TSP decoder attention stack.
// All six GEMM stages run on v_wmma_f32_16x16x32_bf16 (bf16 operands, f32
// accumulate). ~24 GFLOP total, ~150MB traffic -> near the 23.3 TB/s roofline
// once the matmuls are on the matrix pipe.
// ---------------------------------------------------------------------------

#define B_ 32
#define P_ 256
#define N_ 1000
#define NP_ 1024      // N padded to 64 tiles of 16 (32 tiles of 32)
#define E_ 256
#define H_ 16
#define D_ 16

typedef __bf16 bf16x16 __attribute__((ext_vector_type(16)));
typedef float  f32x8   __attribute__((ext_vector_type(8)));
typedef float  f32x4   __attribute__((ext_vector_type(4)));
typedef unsigned int u32x4 __attribute__((ext_vector_type(4)));

union ATile {                 // one WMMA A/B operand: 16 bf16 = 8 VGPRs
    bf16x16 v;
    unsigned short u[16];
    u32x4 q[2];
};
union Pack8 {                 // 8 bf16 packed for one 16B store
    u32x4 q;
    unsigned short u[8];
};

#define WMMA_BF16(A, Bm, C) \
    __builtin_amdgcn_wmma_f32_16x16x32_bf16(false, (A), false, (Bm), (short)0, (C), false, false)

__device__ __forceinline__ unsigned short f2bf(float f) {
    unsigned int u = __float_as_uint(f);
    u += 0x7FFFu + ((u >> 16) & 1u);      // round-to-nearest-even
    return (unsigned short)(u >> 16);
}

// merge lane-local (max,sum) softmax stats across the 16 lanes of a half-wave
__device__ __forceinline__ void merge16(float& m, float& s) {
#pragma unroll
    for (int off = 1; off < 16; off <<= 1) {
        float om = __shfl_xor(m, off, 16);
        float os = __shfl_xor(s, off, 16);
        float nm = fmaxf(m, om);
        s = s * __expf(m - nm) + os * __expf(om - nm);
        m = nm;
    }
}

// --------------------------- stage 0: conversions ---------------------------

__global__ void cvt_bf16_kernel(const float* __restrict__ src,
                                unsigned short* __restrict__ dst, int n) {
    int i = blockIdx.x * blockDim.x + threadIdx.x;
    if (i < n) dst[i] = f2bf(src[i]);
}

// W (E x E, row-major in) -> W^T bf16 (out-major), so WMMA B loads are contiguous
__global__ void wt_kernel(const float* __restrict__ src,
                          unsigned short* __restrict__ dst) {
    int idx = blockIdx.x * blockDim.x + threadIdx.x;   // E*E threads
    int o = idx >> 8, i = idx & 255;
    dst[o * E_ + i] = f2bf(src[i * E_ + o]);
}

// ------------------- stage 1: K/V projections (per head tile) ----------------
// wave = (b, node-tile[64], head). K stored [B,H,NP,D]; V^T stored [B,H,D,NP].

__global__ __launch_bounds__(128) void proj_kv_kernel(
    const unsigned short* __restrict__ encB,
    const unsigned short* __restrict__ WkT,
    const unsigned short* __restrict__ WvT,
    unsigned short* __restrict__ Kb,
    unsigned short* __restrict__ Vt) {
    const int lane = threadIdx.x & 31;
    const int wid  = blockIdx.x * 4 + (threadIdx.x >> 5);
    const int b  = wid >> 10;
    const int nt = (wid >> 4) & 63;
    const int h  = wid & 15;
    const int half = lane >> 4, l16 = lane & 15;
    const int kb8 = half * 8, kb16 = half * 16;

    int nodeA = nt * 16 + l16; if (nodeA > N_ - 1) nodeA = N_ - 1;   // clamp pad
    const u32x4* Arow = (const u32x4*)(encB + ((size_t)b * N_ + nodeA) * E_);
    const u32x4* Bk = (const u32x4*)(WkT + (size_t)(h * 16 + l16) * E_);
    const u32x4* Bv = (const u32x4*)(WvT + (size_t)(h * 16 + l16) * E_);

    f32x8 ck = {0,0,0,0,0,0,0,0}, cv = {0,0,0,0,0,0,0,0};
#pragma unroll
    for (int kc = 0; kc < 8; ++kc) {
        ATile a, bk, bv;
        a.q[0]  = Arow[(kc * 32 + kb16) >> 3];
        a.q[1]  = Arow[(kc * 32 + kb16 + 8) >> 3];
        bk.q[0] = Bk[(kc * 32 + kb16) >> 3];
        bk.q[1] = Bk[(kc * 32 + kb16 + 8) >> 3];
        bv.q[0] = Bv[(kc * 32 + kb16) >> 3];
        bv.q[1] = Bv[(kc * 32 + kb16 + 8) >> 3];
        ck = WMMA_BF16(a.v, bk.v, ck);
        cv = WMMA_BF16(a.v, bv.v, cv);
    }
    const int node0 = nt * 16 + half * 8;
    size_t kidx = (((size_t)b * H_ + h) * NP_ + node0) * D_ + l16;
    Pack8 pv;
#pragma unroll
    for (int j = 0; j < 8; ++j) {
        Kb[kidx + (size_t)j * D_] = f2bf(ck[j]);
        pv.u[j] = f2bf(cv[j]);
    }
    *(u32x4*)(Vt + (((size_t)b * H_ + h) * D_ + l16) * NP_ + node0) = pv.q;
}

// ---------------- stage 2: Q = q1*Wqf + qlast*Wql (bf16 on the fly) ----------

__global__ __launch_bounds__(128) void proj_q_kernel(
    const float* __restrict__ q1, const float* __restrict__ ql,
    const unsigned short* __restrict__ WqfT,
    const unsigned short* __restrict__ WqlT,
    unsigned short* __restrict__ Qb) {
    const int lane = threadIdx.x & 31;
    const int wid  = blockIdx.x * 4 + (threadIdx.x >> 5);
    const int rt = wid >> 4, ct = wid & 15;
    const int half = lane >> 4, l16 = lane & 15;
    const int kb8 = half * 8, kb16 = half * 16;

    const float* A1 = q1 + (size_t)(rt * 16 + l16) * E_;
    const float* A2 = ql + (size_t)(rt * 16 + l16) * E_;
    const u32x4* B1 = (const u32x4*)(WqfT + (size_t)(ct * 16 + l16) * E_);
    const u32x4* B2 = (const u32x4*)(WqlT + (size_t)(ct * 16 + l16) * E_);

    f32x8 c = {0,0,0,0,0,0,0,0};
#pragma unroll
    for (int kc = 0; kc < 8; ++kc) {
        ATile a1, a2, b1, b2;
        const float* p1 = A1 + kc * 32 + kb16;
        const float* p2 = A2 + kc * 32 + kb16;
        f32x4 r0 = *(const f32x4*)(p1),      r1 = *(const f32x4*)(p1 + 4);
        f32x4 r2 = *(const f32x4*)(p1 + 8),  r3 = *(const f32x4*)(p1 + 12);
        f32x4 t0 = *(const f32x4*)(p2),      t1 = *(const f32x4*)(p2 + 4);
        f32x4 t2 = *(const f32x4*)(p2 + 8),  t3 = *(const f32x4*)(p2 + 12);
#pragma unroll
        for (int e = 0; e < 4; ++e) {
            a1.u[e] = f2bf(r0[e]); a1.u[4 + e] = f2bf(r1[e]);
            a1.u[8 + e] = f2bf(r2[e]); a1.u[12 + e] = f2bf(r3[e]);
            a2.u[e] = f2bf(t0[e]); a2.u[4 + e] = f2bf(t1[e]);
            a2.u[8 + e] = f2bf(t2[e]); a2.u[12 + e] = f2bf(t3[e]);
        }
        b1.q[0] = B1[(kc * 32 + kb16) >> 3]; b1.q[1] = B1[(kc * 32 + kb16 + 8) >> 3];
        b2.q[0] = B2[(kc * 32 + kb16) >> 3]; b2.q[1] = B2[(kc * 32 + kb16 + 8) >> 3];
        c = WMMA_BF16(a1.v, b1.v, c);
        c = WMMA_BF16(a2.v, b2.v, c);
    }
#pragma unroll
    for (int j = 0; j < 8; ++j) {
        int row = rt * 16 + j + half * 8;           // flat b*P+p
        int b = row >> 8, p = row & 255;
        Qb[(((size_t)b * H_ + ct) * P_ + p) * D_ + l16] = f2bf(c[j]);
    }
}

// ------------- stage 3: attention, two-pass flash-style softmax --------------
// wave = (b, h, p-tile). K-dim of QK^T is D=16, zero-padded to 32.

__global__ __launch_bounds__(128) void attn_kernel(
    const unsigned short* __restrict__ Qb,
    const unsigned short* __restrict__ Kb,
    const unsigned short* __restrict__ Vt,
    const float* __restrict__ mask,
    unsigned short* __restrict__ AO) {
    __shared__ unsigned short Plds[4][16][32];
    const int wv   = threadIdx.x >> 5;
    const int lane = threadIdx.x & 31;
    const int wid  = blockIdx.x * 4 + wv;
    const int b = wid >> 8, h = (wid >> 4) & 15, pt = wid & 15;
    const int half = lane >> 4, l16 = lane & 15;
    const int kb8 = half * 8, kb16 = half * 16;
    const u32x4 zq = {0, 0, 0, 0};

    ATile aQ;                                              // loop-invariant Q tile
    aQ.q[0] = half ? zq : *(const u32x4*)(Qb + ((((size_t)b * H_ + h) * P_ + pt * 16 + l16) * D_));
    aQ.q[1] = half ? zq : *(const u32x4*)(Qb + ((((size_t)b * H_ + h) * P_ + pt * 16 + l16) * D_ + 8)); // K = 16..31 padding (lane half 1)

    const unsigned short* Kbase = Kb + ((size_t)b * H_ + h) * NP_ * D_;
    const unsigned short* Vbase = Vt + (((size_t)b * H_ + h) * D_ + l16) * NP_;
    const float* Mb = mask + (size_t)b * P_ * N_;

    float m_l[8], s_l[8];
#pragma unroll
    for (int j = 0; j < 8; ++j) { m_l[j] = -1e30f; s_l[j] = 0.f; }

    const f32x8 c0init = {0,0,0,0,0,0,0,0};
    // ---- pass 1: row max / sum (lane-local, merged once) ----
    for (int it = 0; it < 32; ++it) {
        const int n0 = it * 32;
        ATile bk0, bk1;
        if (half == 0) {
            const u32x4* k0 = (const u32x4*)(Kbase + (size_t)(n0 + l16) * D_);
            const u32x4* k1 = (const u32x4*)(Kbase + (size_t)(n0 + 16 + l16) * D_);
            bk0.q[0] = k0[0]; bk0.q[1] = k0[1];
            bk1.q[0] = k1[0]; bk1.q[1] = k1[1];
        } else { bk0.q[0] = zq; bk0.q[1] = zq; bk1.q[0] = zq; bk1.q[1] = zq; }
        f32x8 c0 = WMMA_BF16(aQ.v, bk0.v, c0init);
        f32x8 c1 = WMMA_BF16(aQ.v, bk1.v, c0init);
        const int col0 = n0 + l16, col1 = col0 + 16;
#pragma unroll
        for (int j = 0; j < 8; ++j) {
            const float* mp = Mb + (size_t)(pt * 16 + j + half * 8) * N_;
            float v0 = (col0 < N_) ? c0[j] * 0.25f + mp[col0] : -1e30f;
            float v1 = (col1 < N_) ? c1[j] * 0.25f + mp[col1] : -1e30f;
            float nm = fmaxf(m_l[j], fmaxf(v0, v1));
            s_l[j] = s_l[j] * __expf(m_l[j] - nm) + __expf(v0 - nm) + __expf(v1 - nm);
            m_l[j] = nm;
        }
    }
#pragma unroll
    for (int j = 0; j < 8; ++j) merge16(m_l[j], s_l[j]);

    // ---- pass 2: O = sum_t exp(S_t - M) @ V_t ----
    f32x8 o = {0,0,0,0,0,0,0,0};
    for (int it = 0; it < 32; ++it) {
        const int n0 = it * 32;
        ATile bk0, bk1;
        if (half == 0) {
            const u32x4* k0 = (const u32x4*)(Kbase + (size_t)(n0 + l16) * D_);
            const u32x4* k1 = (const u32x4*)(Kbase + (size_t)(n0 + 16 + l16) * D_);
            bk0.q[0] = k0[0]; bk0.q[1] = k0[1];
            bk1.q[0] = k1[0]; bk1.q[1] = k1[1];
        } else { bk0.q[0] = zq; bk0.q[1] = zq; bk1.q[0] = zq; bk1.q[1] = zq; }
        f32x8 c0 = WMMA_BF16(aQ.v, bk0.v, c0init);
        f32x8 c1 = WMMA_BF16(aQ.v, bk1.v, c0init);
        const int col0 = n0 + l16, col1 = col0 + 16;
#pragma unroll
        for (int j = 0; j < 8; ++j) {
            const float* mp = Mb + (size_t)(pt * 16 + j + half * 8) * N_;
            float v0 = (col0 < N_) ? c0[j] * 0.25f + mp[col0] : -1e30f;
            float v1 = (col1 < N_) ? c1[j] * 0.25f + mp[col1] : -1e30f;
            Plds[wv][j + half * 8][l16]      = f2bf(__expf(v0 - m_l[j]));
            Plds[wv][j + half * 8][16 + l16] = f2bf(__expf(v1 - m_l[j]));
        }
        ATile aP, bV;                   // same-wave LDS ops are in-order (DScnt)
        aP.q[0] = *(const u32x4*)&Plds[wv][l16][kb16];
        aP.q[1] = *(const u32x4*)&Plds[wv][l16][kb16 + 8];
        bV.q[0] = *(const u32x4*)(Vbase + n0 + kb16);
        bV.q[1] = *(const u32x4*)(Vbase + n0 + kb16 + 8);
        o = WMMA_BF16(aP.v, bV.v, o);
    }
#pragma unroll
    for (int j = 0; j < 8; ++j) {
        int p = pt * 16 + j + half * 8;
        AO[((size_t)b * P_ + p) * E_ + h * 16 + l16] = f2bf(o[j] / s_l[j]);
    }
}

// ----------------- stage 4: mh = attn_out @ Wcomb^T + bcomb ------------------

__global__ __launch_bounds__(128) void comb_kernel(
    const unsigned short* __restrict__ AO,
    const unsigned short* __restrict__ WcT,
    const float* __restrict__ bcomb,
    unsigned short* __restrict__ MH) {
    const int lane = threadIdx.x & 31;
    const int wid  = blockIdx.x * 4 + (threadIdx.x >> 5);
    const int rt = wid >> 4, ct = wid & 15;
    const int half = lane >> 4, l16 = lane & 15;
    const int kb8 = half * 8, kb16 = half * 16;
    const int ocol = ct * 16 + l16;

    const u32x4* Arow = (const u32x4*)(AO + (size_t)(rt * 16 + l16) * E_);
    const u32x4* Bw   = (const u32x4*)(WcT + (size_t)ocol * E_);
    f32x8 c = {0,0,0,0,0,0,0,0};
#pragma unroll
    for (int kc = 0; kc < 8; ++kc) {
        ATile a, bw;
        a.q[0]  = Arow[(kc * 32 + kb16) >> 3];
        a.q[1]  = Arow[(kc * 32 + kb16 + 8) >> 3];
        bw.q[0] = Bw[(kc * 32 + kb16) >> 3];
        bw.q[1] = Bw[(kc * 32 + kb16 + 8) >> 3];
        c = WMMA_BF16(a.v, bw.v, c);
    }
    const float bias = bcomb[ocol];
#pragma unroll
    for (int j = 0; j < 8; ++j) {
        int row = rt * 16 + j + half * 8;
        MH[(size_t)row * E_ + ocol] = f2bf(c[j] + bias);
    }
}

// ------ stage 5: softmax(10*tanh(mh @ enc^T / 16) + mask), staged in d_out ---

__global__ __launch_bounds__(128) void final_kernel(
    const unsigned short* __restrict__ MH,
    const unsigned short* __restrict__ encB,
    const float* __restrict__ mask,
    float* __restrict__ out) {
    const int lane = threadIdx.x & 31;
    const int wid  = blockIdx.x * 4 + (threadIdx.x >> 5);
    const int b = wid >> 4, pt = wid & 15;
    const int half = lane >> 4, l16 = lane & 15;
    const int kb8 = half * 8, kb16 = half * 16;

    ATile aM[8];                                  // whole A row set, loop-invariant
    const u32x4* Ar = (const u32x4*)(MH + ((size_t)b * P_ + pt * 16 + l16) * E_);
#pragma unroll
    for (int kc = 0; kc < 8; ++kc) {
        aM[kc].q[0] = Ar[(kc * 32 + kb16) >> 3];
        aM[kc].q[1] = Ar[(kc * 32 + kb16 + 8) >> 3];
    }
    const unsigned short* Eb = encB + (size_t)b * N_ * E_;
    const float* Mb = mask + (size_t)b * P_ * N_;
    float* Ob = out + (size_t)b * P_ * N_;

    float m_l[8], s_l[8];
#pragma unroll
    for (int j = 0; j < 8; ++j) { m_l[j] = -1e30f; s_l[j] = 0.f; }

    for (int nt = 0; nt < 64; ++nt) {
        const int node = nt * 16 + l16;
        const int nodec = (node < N_) ? node : (N_ - 1);
        const u32x4* Brow = (const u32x4*)(Eb + (size_t)nodec * E_);
        f32x8 c = {0,0,0,0,0,0,0,0};
#pragma unroll
        for (int kc = 0; kc < 8; ++kc) {
            ATile bE;
            bE.q[0] = Brow[(kc * 32 + kb16) >> 3];
            bE.q[1] = Brow[(kc * 32 + kb16 + 8) >> 3];
            c = WMMA_BF16(aM[kc].v, bE.v, c);
        }
        if (node < N_) {
#pragma unroll
            for (int j = 0; j < 8; ++j) {
                int p = pt * 16 + j + half * 8;
                float x = c[j] * 0.0625f;               // / SQRT_EMBEDDING_DIM
                float e = __expf(-2.f * fabsf(x));
                float th = copysignf((1.f - e) / (1.f + e), x);
                float l = 10.f * th + Mb[(size_t)p * N_ + node];
                Ob[(size_t)p * N_ + node] = l;          // stage raw logit
                float nm = fmaxf(m_l[j], l);
                s_l[j] = s_l[j] * __expf(m_l[j] - nm) + __expf(l - nm);
                m_l[j] = nm;
            }
        }
    }
#pragma unroll
    for (int j = 0; j < 8; ++j) merge16(m_l[j], s_l[j]);
    float inv[8];
#pragma unroll
    for (int j = 0; j < 8; ++j) inv[j] = 1.f / s_l[j];

    for (int nt = 0; nt < 64; ++nt) {                   // normalize (same-wave RAW
        const int node = nt * 16 + l16;                 // to same addr is ordered)
        if (node < N_) {
#pragma unroll
            for (int j = 0; j < 8; ++j) {
                int p = pt * 16 + j + half * 8;
                float l = Ob[(size_t)p * N_ + node];
                Ob[(size_t)p * N_ + node] = __expf(l - m_l[j]) * inv[j];
            }
        }
    }
}

// --------------------------------- launcher ---------------------------------

extern "C" void kernel_launch(void* const* d_in, const int* in_sizes, int n_in,
                              void* d_out, int out_size, void* d_ws, size_t ws_size,
                              hipStream_t stream) {
    (void)in_sizes; (void)n_in; (void)out_size; (void)ws_size;
    const float* enc   = (const float*)d_in[0];
    const float* q1    = (const float*)d_in[1];
    const float* qlast = (const float*)d_in[2];
    const float* mask  = (const float*)d_in[3];
    const float* Wqf   = (const float*)d_in[4];
    const float* Wql   = (const float*)d_in[5];
    const float* Wk    = (const float*)d_in[6];
    const float* Wv    = (const float*)d_in[7];
    const float* Wc    = (const float*)d_in[8];
    const float* bc    = (const float*)d_in[9];
    float* out = (float*)d_out;

    char* base = (char*)d_ws;
    size_t off = 0;
    auto take = [&](size_t bytes) -> void* {
        void* p = base + off;
        off = (off + bytes + 255) & ~(size_t)255;
        return p;
    };
    unsigned short* encB = (unsigned short*)take((size_t)B_ * N_ * E_ * 2);   // 16.4 MB
    unsigned short* WqfT = (unsigned short*)take((size_t)E_ * E_ * 2);
    unsigned short* WqlT = (unsigned short*)take((size_t)E_ * E_ * 2);
    unsigned short* WkT  = (unsigned short*)take((size_t)E_ * E_ * 2);
    unsigned short* WvT  = (unsigned short*)take((size_t)E_ * E_ * 2);
    unsigned short* WcT  = (unsigned short*)take((size_t)E_ * E_ * 2);
    unsigned short* Kb   = (unsigned short*)take((size_t)B_ * H_ * NP_ * D_ * 2); // 16.8 MB
    unsigned short* Vt   = (unsigned short*)take((size_t)B_ * H_ * D_ * NP_ * 2); // 16.8 MB
    unsigned short* Qb   = (unsigned short*)take((size_t)B_ * H_ * P_ * D_ * 2);
    unsigned short* AO   = (unsigned short*)take((size_t)B_ * P_ * E_ * 2);
    unsigned short* MH   = (unsigned short*)take((size_t)B_ * P_ * E_ * 2);
    // total workspace ~63 MB

    const int nEnc = B_ * N_ * E_;
    cvt_bf16_kernel<<<(nEnc + 255) / 256, 256, 0, stream>>>(enc, encB, nEnc);
    wt_kernel<<<E_ * E_ / 256, 256, 0, stream>>>(Wqf, WqfT);
    wt_kernel<<<E_ * E_ / 256, 256, 0, stream>>>(Wql, WqlT);
    wt_kernel<<<E_ * E_ / 256, 256, 0, stream>>>(Wk,  WkT);
    wt_kernel<<<E_ * E_ / 256, 256, 0, stream>>>(Wv,  WvT);
    wt_kernel<<<E_ * E_ / 256, 256, 0, stream>>>(Wc,  WcT);

    proj_kv_kernel<<<B_ * 64 * H_ / 4, 128, 0, stream>>>(encB, WkT, WvT, Kb, Vt);
    proj_q_kernel<<<(B_ * P_ / 16) * (E_ / 16) / 4, 128, 0, stream>>>(q1, qlast, WqfT, WqlT, Qb);
    attn_kernel<<<B_ * H_ * (P_ / 16) / 4, 128, 0, stream>>>(Qb, Kb, Vt, mask, AO);
    comb_kernel<<<(B_ * P_ / 16) * (E_ / 16) / 4, 128, 0, stream>>>(AO, WcT, bc, MH);
    final_kernel<<<B_ * (P_ / 16) / 4, 128, 0, stream>>>(MH, encB, mask, out);
}
